// APPNPLinear_66288525246941
// MI455X (gfx1250) — matrix operand order repaired and checked by reading, not compiled, not run on hardware
//
#include <hip/hip_runtime.h>
#include <stdint.h>

#define ALPHA 0.1f
#define BETA  0.9f
#define D_IN  128
#define D_OUT 64
#define K_STEPS 10

typedef __attribute__((ext_vector_type(4)))  __bf16 v4bf;   // 8 bytes
typedef __attribute__((ext_vector_type(16))) __bf16 v16bf;  // 32 bytes
typedef __attribute__((ext_vector_type(8)))  float  v8f;

struct B256 { uint4 lo, hi; };  // 32 bytes, for bit_cast to v16bf

// ---------------- degree / normalization ----------------
__global__ void deg_init_kernel(float* __restrict__ deg, int N) {
    int i = blockIdx.x * blockDim.x + threadIdx.x;
    if (i < N) deg[i] = 1.0f;  // self-loop
}

__global__ void deg_edges_kernel(const int* __restrict__ dst,
                                 float* __restrict__ deg, long long E) {
    long long e = (long long)blockIdx.x * blockDim.x + threadIdx.x;
    if (e < E) atomicAdd(&deg[dst[e]], 1.0f);
}

__global__ void deg_rsqrt_kernel(float* __restrict__ deg, int N) {
    int i = blockIdx.x * blockDim.x + threadIdx.x;
    if (i < N) deg[i] = rsqrtf(deg[i]);  // in place -> dinv
}

// ------------- per-iteration init: nxt = alpha*x + beta*dinv^2*cur -------------
__global__ __launch_bounds__(256) void appnp_prep_kernel(
    const float* __restrict__ x, const float* __restrict__ cur,
    const float* __restrict__ dinv, float* __restrict__ nxt, int N) {
    int i = blockIdx.x * blockDim.x + threadIdx.x;   // over N*32 float4 groups
    if (i >= N * (D_IN / 4)) return;
    int n = i >> 5;
    float di = dinv[n];
    float c = BETA * di * di;
    float4 xv = ((const float4*)x)[i];
    float4 hv = ((const float4*)cur)[i];
    float4 r;
    r.x = ALPHA * xv.x + c * hv.x;
    r.y = ALPHA * xv.y + c * hv.y;
    r.z = ALPHA * xv.z + c * hv.z;
    r.w = ALPHA * xv.w + c * hv.w;
    ((float4*)nxt)[i] = r;
}

// ------------- edge scatter: one wave per edge, float4 per lane -------------
__global__ __launch_bounds__(256) void appnp_edge_kernel(
    const int* __restrict__ src, const int* __restrict__ dst,
    const float* __restrict__ dinv, const float* __restrict__ cur,
    float* __restrict__ nxt, long long E) {
    long long wave = (long long)blockIdx.x * (blockDim.x / 32) + (threadIdx.x >> 5);
    if (wave >= E) return;
    int lane = threadIdx.x & 31;
    int s = src[wave];
    int t = dst[wave];
    float coef = BETA * dinv[s] * dinv[t];
    float4 v = ((const float4*)(cur + (size_t)s * D_IN))[lane];
    float* o = nxt + (size_t)t * D_IN + lane * 4;
    atomicAdd(o + 0, coef * v.x);
    atomicAdd(o + 1, coef * v.y);
    atomicAdd(o + 2, coef * v.z);
    atomicAdd(o + 3, coef * v.w);
}

// ------------- h (f32, row-major) -> h (bf16, row-major) -------------
__global__ __launch_bounds__(256) void cvt_h_bf16_kernel(
    const float* __restrict__ h, __bf16* __restrict__ hb, int n4) {
    int i = blockIdx.x * blockDim.x + threadIdx.x;
    if (i >= n4) return;
    float4 v = ((const float4*)h)[i];
    v4bf r;
    r[0] = (__bf16)v.x;
    r[1] = (__bf16)v.y;
    r[2] = (__bf16)v.z;
    r[3] = (__bf16)v.w;
    ((v4bf*)hb)[i] = r;
}

// ------------- pre-pack W (128x64 f32) into per-lane WMMA B-fragment layout -------------
// fragment f = kb*4 + nt; per lane 16 bf16 contiguous (32B):
//   col = n0 + lane%16 ; K = kb*32 + (lane/16)*8 + 2j + (j>=4 ? 8 : 0), +1  (j = 0..7)
__global__ void pack_w_kernel(const float* __restrict__ W, __bf16* __restrict__ wp) {
    int t = blockIdx.x * blockDim.x + threadIdx.x;
    if (t >= 512) return;             // 16 fragments * 32 lanes
    int frag = t >> 5;
    int lane = t & 31;
    int kb = frag >> 2, nt = frag & 3;
    int half = lane >> 4, lm = lane & 15;
    int kbase = kb * 32 + half * 8;
    int n0 = nt * 16;
    __bf16* d = wp + ((size_t)frag * 32 + lane) * 16;
#pragma unroll
    for (int j = 0; j < 8; ++j) {
        d[2 * j]     = (__bf16)W[(size_t)(kbase + 2 * j + (j >= 4 ? 8 : 0)) * D_OUT + n0 + lm];
        d[2 * j + 1] = (__bf16)W[(size_t)(kbase + 2 * j + 1 + (j >= 4 ? 8 : 0)) * D_OUT + n0 + lm];
    }
}

// ------------- final GEMM: out = h @ W + b via v_wmma_f32_16x16x32_bf16 -------------
// One wave computes a 16x64 tile: 4 k-chunks x 4 n-tiles, all operands via b128 loads.
__global__ __launch_bounds__(256) void appnp_gemm_wmma_kernel(
    const __bf16* __restrict__ hb, const __bf16* __restrict__ wp,
    const float* __restrict__ bias, float* __restrict__ out, int N) {
    int wave = (blockIdx.x * (blockDim.x / 32)) + (threadIdx.x >> 5);
    int lane = threadIdx.x & 31;
    int m0 = wave * 16;
    if (m0 >= N) return;                 // wave-uniform: EXEC all-1 inside

    int half = lane >> 4;
    int lm   = lane & 15;
    int row  = m0 + lm;

    v8f acc[4];
    v8f zero = {0.f, 0.f, 0.f, 0.f, 0.f, 0.f, 0.f, 0.f};
#pragma unroll
    for (int nt = 0; nt < 4; ++nt) acc[nt] = zero;

    // row of bf16 h viewed as dwords: dword d holds K=2d (lo), K=2d+1 (hi)
    const uint* hrow = (const uint*)(hb + (size_t)row * D_IN);

#pragma unroll
    for (int kb = 0; kb < 4; ++kb) {
        // A fragment: dwords [kb*16 + half*4 + 0..3] and [kb*16 + 8 + half*4 + 0..3]
        const uint4* pa = (const uint4*)(hrow + kb * 16 + half * 4);
        B256 araw;
        araw.lo = pa[0];
        araw.hi = pa[2];                 // +8 dwords
        v16bf a = __builtin_bit_cast(v16bf, araw);
#pragma unroll
        for (int nt = 0; nt < 4; ++nt) {
            const uint4* pb =
                (const uint4*)(wp + ((size_t)(kb * 4 + nt) * 32 + lane) * 16);
            B256 braw;
            braw.lo = pb[0];
            braw.hi = pb[1];
            v16bf b = __builtin_bit_cast(v16bf, braw);
            acc[nt] = __builtin_amdgcn_wmma_f32_16x16x32_bf16(
                false, a, false, b, (short)0, acc[nt], false, false);
        }
    }

    // store C (f32 16x16 layout: VGPR r -> M = m0 + half*8 + r, N = n0 + lm)
#pragma unroll
    for (int nt = 0; nt < 4; ++nt) {
        int n = nt * 16 + lm;
        float bv = bias[n];
#pragma unroll
        for (int r = 0; r < 8; ++r) {
            int m = m0 + half * 8 + r;
            out[(size_t)m * D_OUT + n] = acc[nt][r] + bv;
        }
    }
}

extern "C" void kernel_launch(void* const* d_in, const int* in_sizes, int n_in,
                              void* d_out, int out_size, void* d_ws, size_t ws_size,
                              hipStream_t stream) {
    const float* x    = (const float*)d_in[0];
    const int*   ei   = (const int*)d_in[1];     // edge_index flat [2, E]
    const float* W    = (const float*)d_in[2];
    const float* bias = (const float*)d_in[3];
    float*       out  = (float*)d_out;

    int N = in_sizes[0] / D_IN;
    long long E = (long long)in_sizes[1] / 2;
    const int* src = ei;
    const int* dst = ei + E;

    // workspace layout: dinv[N] | Wpack[8192 bf16] | hA[N*128 f32] | hB[N*128 f32]
    char* ws = (char*)d_ws;
    float* dinv = (float*)ws;
    size_t off = ((size_t)N * sizeof(float) + 255) & ~(size_t)255;
    __bf16* wpack = (__bf16*)(ws + off);
    off += (16 * 32 * 16 * sizeof(__bf16) + 255) & ~(size_t)255;
    float* hA = (float*)(ws + off);
    float* hB = (float*)(ws + off + (size_t)N * D_IN * sizeof(float));

    // 1) degree -> dinv ; pack W fragments (independent)
    deg_init_kernel<<<(N + 255) / 256, 256, 0, stream>>>(dinv, N);
    deg_edges_kernel<<<(unsigned)((E + 255) / 256), 256, 0, stream>>>(dst, dinv, E);
    deg_rsqrt_kernel<<<(N + 255) / 256, 256, 0, stream>>>(dinv, N);
    pack_w_kernel<<<1, 512, 0, stream>>>(W, wpack);

    // 2) K propagation steps (ping-pong; iteration 1 reads x directly)
    const float* cur = x;
    float* bufs[2] = {hA, hB};
    int which = 0;
    int prepBlocks = (N * (D_IN / 4) + 255) / 256;
    long long wavesPerBlock = 256 / 32;
    unsigned edgeBlocks = (unsigned)((E + wavesPerBlock - 1) / wavesPerBlock);
    for (int it = 0; it < K_STEPS; ++it) {
        float* nxt = bufs[which];
        appnp_prep_kernel<<<prepBlocks, 256, 0, stream>>>(x, cur, dinv, nxt, N);
        appnp_edge_kernel<<<edgeBlocks, 256, 0, stream>>>(src, dst, dinv, cur, nxt, E);
        cur = nxt;
        which ^= 1;
    }

    // 3) convert final h to bf16 into the now-free ping-pong buffer
    __bf16* hbf = (__bf16*)bufs[which];   // free buffer (51.2MB >= 25.6MB needed)
    int n4 = N * (D_IN / 4);
    cvt_h_bf16_kernel<<<(n4 + 255) / 256, 256, 0, stream>>>(cur, hbf, n4);

    // 4) final linear via WMMA
    int mtiles = (N + 15) / 16;
    int gemmBlocks = (mtiles + 7) / 8;   // 8 waves per 256-thread block
    appnp_gemm_wmma_kernel<<<gemmBlocks, 256, 0, stream>>>(hbf, wpack, bias, out, N);
}
